// Int4Conv1D_90752658964993
// MI455X (gfx1250) — compile-verified, matrix operand order checked by reading, not run on hardware
//
#include <hip/hip_runtime.h>
#include <hip/hip_bf16.h>

// ---------------------------------------------------------------------------
// Int4-dequant GEMM:  out[M,N] = x[M,K] * W[N,K]^T + bias,  W = (q - z) * s
//   M = 4*2048 = 8192, K = NX = 2048, N = NF = 8192, GROUP = 64
// f16 WMMA (v_wmma_f32_16x16x32_f16) with f32 accumulation.
// ---------------------------------------------------------------------------

typedef __attribute__((ext_vector_type(16))) _Float16 v16h;
typedef __attribute__((ext_vector_type(8)))  _Float16 v8h;
typedef __attribute__((ext_vector_type(8)))  float    v8f;
typedef __attribute__((ext_vector_type(4)))  float    v4f;
typedef __attribute__((ext_vector_type(4)))  int      v4i;

#define NX_C    2048
#define NF_C    8192
#define MTOT_C  8192
#define NG_C    32          // NX / 64 groups
#define BM      128
#define BN      128
#define BK      32
#define LDSK    40          // padded row stride in halves (80B -> conflict-free b128 reads)

// Per-thread staging registers for one K-step of global traffic.
struct Stage {
    v4f x0, x1, x2, x3;     // 16 floats of x
    v4i p0, p1;             // 8 packed int32 bytes -> 16 int4 weights
    float s;                // group scale for this feature
    int   z;                // group zero  for this feature
};

__device__ __forceinline__ Stage stage_load(const float* __restrict__ xrow,
                                            const int*   __restrict__ prow,
                                            const float* __restrict__ srow,
                                            const int*   __restrict__ zrow,
                                            int kk, int lhalf) {
    Stage st;
    const float* xp = xrow + kk + lhalf * 16;
    st.x0 = *(const v4f*)(xp + 0);
    st.x1 = *(const v4f*)(xp + 4);
    st.x2 = *(const v4f*)(xp + 8);
    st.x3 = *(const v4f*)(xp + 12);
    const int* pp = prow + (kk >> 1) + lhalf * 8;
    st.p0 = *(const v4i*)(pp + 0);
    st.p1 = *(const v4i*)(pp + 4);
    const int g = kk >> 6;           // GROUP_SIZE = 64, BK = 32 -> one group per step
    st.s = srow[g];
    st.z = zrow[g];
    return st;
}

__device__ __forceinline__ void stage_commit(const Stage& st,
                                             _Float16* __restrict__ As,
                                             _Float16* __restrict__ Bs,
                                             int lrow, int lhalf) {
    // ---- x tile: f32 -> f16, two 16B LDS stores ----
    v8h a0, a1;
    #pragma unroll
    for (int i = 0; i < 4; ++i) {
        a0[i]     = (_Float16)st.x0[i];
        a0[i + 4] = (_Float16)st.x1[i];
        a1[i]     = (_Float16)st.x2[i];
        a1[i + 4] = (_Float16)st.x3[i];
    }
    _Float16* ap = As + lrow * LDSK + lhalf * 16;
    *(v8h*)(ap + 0) = a0;
    *(v8h*)(ap + 8) = a1;

    // ---- W tile: nibble dequant -> f16, two 16B LDS stores ----
    const float s = st.s;
    const int   z = st.z;
    v8h b0, b1;
    #pragma unroll
    for (int i = 0; i < 4; ++i) {
        int by = st.p0[i];
        b0[2 * i]     = (_Float16)((float)((by & 15) - z) * s);
        b0[2 * i + 1] = (_Float16)((float)(((by >> 4) & 15) - z) * s);
        by = st.p1[i];
        b1[2 * i]     = (_Float16)((float)((by & 15) - z) * s);
        b1[2 * i + 1] = (_Float16)((float)(((by >> 4) & 15) - z) * s);
    }
    _Float16* bp = Bs + lrow * LDSK + lhalf * 16;
    *(v8h*)(bp + 0) = b0;
    *(v8h*)(bp + 8) = b1;
}

// Load one 16x32 f16 WMMA operand fragment from LDS (ISA 7.12.2 layout):
//   lane<16 : row = lane,    K = {0..7, 16..23}
//   lane>=16: row = lane-16, K = {8..15, 24..31}
__device__ __forceinline__ v16h frag_ld(const _Float16* __restrict__ base,
                                        int row_base, int lane) {
    const _Float16* p = base + (row_base + (lane & 15)) * LDSK + ((lane >> 4) << 3);
    v8h lo = *(const v8h*)(p);        // K = b .. b+7
    v8h hi = *(const v8h*)(p + 16);   // K = b+16 .. b+23
    v16h r;
    #pragma unroll
    for (int i = 0; i < 8; ++i) { r[i] = lo[i]; r[i + 8] = hi[i]; }
    return r;
}

__global__ __launch_bounds__(256, 2)
void int4_gemm_wmma(const float* __restrict__ x,      // [MTOT, NX]
                    const int*   __restrict__ packed, // [NF, NX/2]
                    const float* __restrict__ scales, // [NF, NG]
                    const int*   __restrict__ zeros,  // [NF, NG]
                    const float* __restrict__ bias,   // [NF]
                    float*       __restrict__ out) {  // [MTOT, NF]
    __shared__ __align__(16) _Float16 As[BM * LDSK];  // x   tile, f16
    __shared__ __align__(16) _Float16 Bs[BN * LDSK];  // W   tile, f16 (dequantized)

    const int tid  = threadIdx.x;
    const int lane = tid & 31;
    const int wave = tid >> 5;                 // 0..7
    const int blockMBase = blockIdx.y * BM;
    const int blockNBase = blockIdx.x * BN;

    // wave grid 4(M) x 2(N): each wave -> 32x64 output = 2x4 WMMA tiles
    const int wm = (wave >> 1) * 32;
    const int wn = (wave & 1) * 64;

    // cooperative loader mapping: 2 threads per row, 16 K-elems each
    const int lrow  = tid >> 1;                // 0..127
    const int lhalf = tid & 1;                 // 0/1

    const float* xrow = x      + (size_t)(blockMBase + lrow) * NX_C;
    const int*   prow = packed + (size_t)(blockNBase + lrow) * (NX_C / 2);
    const float* srow = scales + (size_t)(blockNBase + lrow) * NG_C;
    const int*   zrow = zeros  + (size_t)(blockNBase + lrow) * NG_C;

    v8f acc[2][4] = {};

    // -------- software-pipelined K loop --------
    Stage st = stage_load(xrow, prow, srow, zrow, 0, lhalf);

    for (int kk = 0; kk < NX_C; kk += BK) {
        stage_commit(st, As, Bs, lrow, lhalf);
        __syncthreads();

        if (kk + BK < NX_C)
            st = stage_load(xrow, prow, srow, zrow, kk + BK, lhalf);

        v16h aF[2], bF[4];
        #pragma unroll
        for (int mi = 0; mi < 2; ++mi) aF[mi] = frag_ld(As, wm + mi * 16, lane);
        #pragma unroll
        for (int ni = 0; ni < 4; ++ni) bF[ni] = frag_ld(Bs, wn + ni * 16, lane);

        #pragma unroll
        for (int mi = 0; mi < 2; ++mi)
            #pragma unroll
            for (int ni = 0; ni < 4; ++ni)
                acc[mi][ni] = __builtin_amdgcn_wmma_f32_16x16x32_f16(
                    /*neg_a=*/false, aF[mi], /*neg_b=*/false, bF[ni],
                    /*c_mod=*/(short)0, acc[mi][ni],
                    /*reuse_a=*/false, /*reuse_b=*/false);

        __syncthreads();
    }

    // -------- epilogue: bias add + store (C/D layout: VGPR j -> M = j + 8*(lane>=16)) --------
    const int halfsel = (lane >> 4) << 3;
    const int ncol    = lane & 15;
    #pragma unroll
    for (int ni = 0; ni < 4; ++ni) {
        const int n  = blockNBase + wn + ni * 16 + ncol;
        const float bv = bias[n];
        #pragma unroll
        for (int mi = 0; mi < 2; ++mi) {
            #pragma unroll
            for (int j = 0; j < 8; ++j) {
                const int m = blockMBase + wm + mi * 16 + j + halfsel;
                out[(size_t)m * NF_C + n] = acc[mi][ni][j] + bv;
            }
        }
    }
}

extern "C" void kernel_launch(void* const* d_in, const int* in_sizes, int n_in,
                              void* d_out, int out_size, void* d_ws, size_t ws_size,
                              hipStream_t stream) {
    (void)in_sizes; (void)n_in; (void)out_size; (void)d_ws; (void)ws_size;
    const float* x      = (const float*)d_in[0];
    const int*   packed = (const int*)  d_in[1];
    const float* scales = (const float*)d_in[2];
    const int*   zeros  = (const int*)  d_in[3];
    const float* bias   = (const float*)d_in[4];
    float*       out    = (float*)d_out;

    dim3 grid(NF_C / BN, MTOT_C / BM);   // 64 x 64
    dim3 block(256);
    int4_gemm_wmma<<<grid, block, 0, stream>>>(x, packed, scales, zeros, bias, out);
}
